// T12_5557687681830
// MI455X (gfx1250) — matrix operand order repaired and checked by reading, not compiled, not run on hardware
//
#include <hip/hip_runtime.h>

// ---------------------------------------------------------------------------
// MI455X (gfx1250) fused BN + Linear/ReLU + concat + Linear
//   out[n,g] = sum_k cat[n,k] * w2[g,k],  cat = [h, relu(bn(agg) @ w1^T)]
// BN folded into w1. f16 WMMA 16x16x32, fp32 accumulate. Weights live in the
// 320KB WGP LDS (bank-conflict-padded rows). Each wave owns TWO 16-row
// stripes so every B fragment read from LDS feeds two WMMAs (halves LDS
// traffic per FLOP -> stays under the LDS bandwidth ceiling at the HBM-bound
// rate of ~1 WMMA / 4 cycles / WGP).
// ---------------------------------------------------------------------------

typedef __attribute__((ext_vector_type(16))) _Float16 v16h;
typedef __attribute__((ext_vector_type(8)))  _Float16 v8h;
typedef __attribute__((ext_vector_type(8)))  float    v8f;

#define PREVF 127
#define AGGF  128
#define OUTF  255
#define NBLK_STATS 512
#define WROW 136   // padded LDS row stride in halfs (68 dwords) -> conflict-free B reads

static __device__ __forceinline__ v8f wmma_f16(v16h a, v16h b, v8f c) {
  // D = A(16x32 f16) * B(32x16 f16) + C(16x16 f32)
  return __builtin_amdgcn_wmma_f32_16x16x32_f16(false, a, false, b, (short)0, c,
                                                false, false);
}

static __device__ __forceinline__ v16h cat8(v8h a, v8h b) {
  return __builtin_shufflevector(a, b, 0, 1, 2, 3, 4, 5, 6, 7,
                                       8, 9, 10, 11, 12, 13, 14, 15);
}

// B fragment: lane L holds column N = L%16, K = kbase + (L/16)*8 + j (j<8) and
// kbase + 16 + (L/16)*8 + (j-8). Row-major [g][k] LDS with WROW stride -> two 16B loads.
static __device__ __forceinline__ v16h ldsB(const _Float16* p) {
  v8h b0 = *(const v8h*)p;
  v8h b1 = *(const v8h*)(p + 16);
  return cat8(b0, b1);
}

// ---------------- Pass 1: deterministic column sums of agg ----------------
__global__ void k_stats(const float* __restrict__ agg, float* __restrict__ psum,
                        float* __restrict__ psq, int N, int rpb) {
  const int col = threadIdx.x;            // 128 threads
  const long r0 = (long)blockIdx.x * rpb;
  float s = 0.f, q = 0.f;
  for (int i = 0; i < rpb; ++i) {
    const long r = r0 + i;
    if (r >= N) break;
    const float v = agg[r * AGGF + col];
    s += v; q += v * v;
  }
  psum[blockIdx.x * AGGF + col] = s;
  psq [blockIdx.x * AGGF + col] = q;
}

__global__ void k_finalize(const float* __restrict__ psum,
                           const float* __restrict__ psq,
                           float* __restrict__ mean, float* __restrict__ inv,
                           int N) {
  const int col = threadIdx.x;            // 128 threads
  float s = 0.f, q = 0.f;
  for (int b = 0; b < NBLK_STATS; ++b) {
    s += psum[b * AGGF + col];
    q += psq [b * AGGF + col];
  }
  const float m   = s / (float)N;
  const float var = q / (float)N - m * m;
  mean[col] = m;
  inv[col]  = rsqrtf(var + 1e-5f);
}

// ---------------- Pass 2: fold BN into w1, repack weights to f16 ----------
// wq layout (halfs): [0,16384) w1'  [16384,49152) w2 h-part  [49152,81920) w2 a-part
__global__ void k_prep(const float* __restrict__ w1, const float* __restrict__ w2,
                       const float* __restrict__ mean, const float* __restrict__ inv,
                       float* __restrict__ bias1, _Float16* __restrict__ wq) {
  __shared__ float red[128];
  const int k = threadIdx.x;              // 128 threads
  const int g = blockIdx.x;               // 384 blocks
  if (g < 128) {
    const float t = w1[g * AGGF + k] * inv[k];
    wq[g * 128 + k] = (_Float16)t;
    red[k] = t * mean[k];
    __syncthreads();
    for (int off = 64; off > 0; off >>= 1) {
      if (k < off) red[k] += red[k + off];
      __syncthreads();
    }
    if (k == 0) bias1[g] = -red[0];
  } else {
    const int gg = g - 128;               // 0..255 (row 255 = zero pad)
    float vh = 0.f, va = 0.f;
    if (gg < OUTF) {
      if (k < PREVF) vh = w2[gg * OUTF + k];       // cat k in [0,127)  = h
      va = w2[gg * OUTF + PREVF + k];              // cat k in [127,255) = a
    }
    wq[16384 + gg * 128 + k] = (_Float16)vh;
    wq[49152 + gg * 128 + k] = (_Float16)va;
  }
}

// ---------------- Pass 3: fused dual-GEMM over 256-row tiles --------------
// 8 waves * 2 stripes * 16 rows = 256 rows per CTA iteration.
__global__ __launch_bounds__(256, 1)
void k_fused(const float* __restrict__ hin, const float* __restrict__ agg,
             float* __restrict__ out, const _Float16* __restrict__ wq,
             const float* __restrict__ bias1, int N, int nTiles) {
  __shared__ alignas(16) _Float16 s_w[640 * WROW];       // 174,080 B of weights
  __shared__ alignas(16) _Float16 s_a[8 * 32 * WROW];    // per-wave 32-row a-tiles
  __shared__ float s_bias[128];

  const int tid = threadIdx.x;
  // Stage weights: 640 rows of 128 halfs -> padded rows of WROW halfs.
  for (int i = tid; i < 640 * 8; i += 256) {
    const int row = i >> 3, c = i & 7;
    ((uint4*)(s_w + row * WROW))[c] = ((const uint4*)(wq + row * 128))[c];
  }
  if (tid < 128) s_bias[tid] = bias1[tid];
  __syncthreads();

  const int wave = tid >> 5, lane = tid & 31;
  const int lhi = lane >> 4, llo = lane & 15;
  _Float16* SA = s_a + wave * 32 * WROW;                 // two 16-row tiles
  const _Float16* W1  = s_w;
  const _Float16* W2H = s_w + 128 * WROW;
  const _Float16* W2A = s_w + 384 * WROW;

  for (long tile = blockIdx.x; tile < nTiles; tile += gridDim.x) {
    const long r0 = tile * 256 + wave * 32;              // this wave's 32 rows
    long rowA[2];
    #pragma unroll
    for (int m = 0; m < 2; ++m) {
      rowA[m] = r0 + m * 16 + llo;                       // A row = lane%16
      if (rowA[m] >= N) rowA[m] = N - 1;                 // clamp; never stored
    }

    // Prefetch next tile's activation rows toward this WGP (global_prefetch).
    {
      const long nt = tile + gridDim.x;
      if (nt < nTiles) {
        long pr = nt * 256 + wave * 32 + llo + lhi * 16;
        if (pr >= N) pr = N - 1;
        __builtin_prefetch(agg + pr * AGGF, 0, 3);
        __builtin_prefetch(hin + pr * PREVF, 0, 3);
      }
    }

    // ---- A1 fragments from agg (fp32 -> f16), K = 128, 2 stripes --------
    // 16-bit A layout: half j holds K = 32s + (j/8)*16 + (lane/16)*8 + j%8.
    v16h A1[2][4];
    #pragma unroll
    for (int m = 0; m < 2; ++m) {
      const float* ap = agg + rowA[m] * AGGF;
      #pragma unroll
      for (int s = 0; s < 4; ++s) {
        const int c0 = s * 32 + lhi * 8;
        const float* p0 = ap + c0;
        const float* p1 = ap + c0 + 16;
        v16h a;
        #pragma unroll
        for (int j = 0; j < 8; ++j) {
          a[j]     = (_Float16)p0[j];
          a[8 + j] = (_Float16)p1[j];
        }
        A1[m][s] = a;
      }
    }

    // ---- GEMM1: a = relu(agg @ w1'^T + bias1); a-tiles -> LDS -----------
    // Each B fragment feeds both stripes (2 WMMAs per LDS read).
    #pragma unroll
    for (int n1 = 0; n1 < 8; ++n1) {
      const int g = n1 * 16 + llo;                       // output feature
      const float b = s_bias[g];
      v8f acc0, acc1;
      #pragma unroll
      for (int r = 0; r < 8; ++r) { acc0[r] = b; acc1[r] = b; }
      const _Float16* wrow = W1 + g * WROW + lhi * 8;
      #pragma unroll
      for (int s = 0; s < 4; ++s) {
        const v16h B = ldsB(wrow + s * 32);
        acc0 = wmma_f16(A1[0][s], B, acc0);
        acc1 = wmma_f16(A1[1][s], B, acc1);
      }
      // C layout: VGPR r -> row M = r + 8*(lane/16), col N = lane%16
      #pragma unroll
      for (int r = 0; r < 8; ++r) {
        float v0 = acc0[r]; v0 = v0 > 0.f ? v0 : 0.f;
        float v1 = acc1[r]; v1 = v1 > 0.f ? v1 : 0.f;
        SA[(r + 8 * lhi) * WROW + g]              = (_Float16)v0;
        SA[(16 + r + 8 * lhi) * WROW + g]         = (_Float16)v1;
      }
    }

    // ---- A2 fragments: h part (global fp32, k=127 padded) + a part (LDS)-
    v16h A2h[2][4], A2a[2][4];
    #pragma unroll
    for (int m = 0; m < 2; ++m) {
      const float* hp = hin + rowA[m] * PREVF;
      #pragma unroll
      for (int s = 0; s < 4; ++s) {
        const int c0 = s * 32 + lhi * 8;
        v16h a;
        #pragma unroll
        for (int j = 0; j < 8; ++j) {
          const int k0 = c0 + j, k1 = c0 + 16 + j;
          a[j]     = (k0 < PREVF) ? (_Float16)hp[k0] : (_Float16)0.f;
          a[8 + j] = (k1 < PREVF) ? (_Float16)hp[k1] : (_Float16)0.f;
        }
        A2h[m][s] = a;
      }
      const _Float16* sp = SA + (m * 16 + llo) * WROW;   // row M = lane%16
      #pragma unroll
      for (int s = 0; s < 4; ++s) {
        const int c0 = s * 32 + lhi * 8;
        v8h a0 = *(const v8h*)(sp + c0);
        v8h a1 = *(const v8h*)(sp + c0 + 16);
        A2a[m][s] = cat8(a0, a1);
      }
    }

    // ---- GEMM2: out = h @ w2h^T + a @ w2a^T; B reused across stripes ----
    for (int n2 = 0; n2 < 16; ++n2) {
      const int col = n2 * 16 + llo;
      v8f acc0 = {}, acc1 = {};
      const _Float16* wh = W2H + col * WROW + lhi * 8;
      const _Float16* wa = W2A + col * WROW + lhi * 8;
      #pragma unroll
      for (int s = 0; s < 4; ++s) {
        const v16h B = ldsB(wh + s * 32);
        acc0 = wmma_f16(A2h[0][s], B, acc0);
        acc1 = wmma_f16(A2h[1][s], B, acc1);
      }
      #pragma unroll
      for (int s = 0; s < 4; ++s) {
        const v16h B = ldsB(wa + s * 32);
        acc0 = wmma_f16(A2a[0][s], B, acc0);
        acc1 = wmma_f16(A2a[1][s], B, acc1);
      }
      if (col < OUTF) {
        #pragma unroll
        for (int r = 0; r < 8; ++r) {
          const long row0 = r0 + r + 8 * lhi;
          const long row1 = r0 + 16 + r + 8 * lhi;
          if (row0 < N) out[row0 * OUTF + col] = acc0[r];
          if (row1 < N) out[row1 * OUTF + col] = acc1[r];
        }
      }
    }
  }
}

// ---------------------------------------------------------------------------
extern "C" void kernel_launch(void* const* d_in, const int* in_sizes, int n_in,
                              void* d_out, int out_size, void* d_ws, size_t ws_size,
                              hipStream_t stream) {
  (void)n_in; (void)out_size; (void)ws_size;
  const float* h   = (const float*)d_in[0];
  const float* agg = (const float*)d_in[1];
  const float* w1  = (const float*)d_in[2];
  const float* w2  = (const float*)d_in[3];
  float* out = (float*)d_out;
  const int N = in_sizes[0] / PREVF;

  // Workspace layout (floats):
  float* ws    = (float*)d_ws;
  float* psum  = ws;                                   // 512*128
  float* psq   = ws + NBLK_STATS * AGGF;               // 512*128
  float* mean  = ws + 2 * NBLK_STATS * AGGF;           // 128
  float* inv   = mean + AGGF;                          // 128
  float* bias1 = inv + AGGF;                           // 128
  _Float16* wq = (_Float16*)(bias1 + AGGF);            // 81920 halfs (16B aligned)

  const int rpb = (N + NBLK_STATS - 1) / NBLK_STATS;
  k_stats<<<NBLK_STATS, 128, 0, stream>>>(agg, psum, psq, N, rpb);
  k_finalize<<<1, 128, 0, stream>>>(psum, psq, mean, inv, N);
  k_prep<<<384, 128, 0, stream>>>(w1, w2, mean, inv, bias1, wq);

  const int nTiles = (N + 255) / 256;
  const int grid = nTiles < 640 ? nTiles : 640;
  k_fused<<<grid, 256, 0, stream>>>(h, agg, out, wq, bias1, N, nTiles);
}
